// JiTBlock_52536039965183
// MI455X (gfx1250) — matrix operand order repaired and checked by reading, not compiled, not run on hardware
//
#include <hip/hip_runtime.h>

// Problem dims
#define BDIM   4
#define NSEQ   1024
#define CDIM   1024
#define HNUM   16
#define DDIM   64
#define BN_TOK 4096          // B*N
#define FFH    2730
#define FF2    5460          // 2*FFH
#define FF2P   5504          // 2*FFH padded to 128
#define FFHP   2752          // FFH padded to 32 (86*32)
#define ADA6   6144          // 6*C

typedef __attribute__((ext_vector_type(16))) __bf16 v16bf;
typedef __attribute__((ext_vector_type(8)))  float  v8f;

__device__ inline unsigned short f2bf(float f) {
  unsigned int u = __float_as_uint(f);
  u += 0x7FFFu + ((u >> 16) & 1u);               // round-to-nearest-even
  return (unsigned short)(u >> 16);
}
__device__ inline float bf2f(unsigned short h) {
  return __uint_as_float(((unsigned int)h) << 16);
}

union FragU { v16bf v; uint4 u[2]; };

// A-matrix 16x32 bf16 fragment (ISA 7.12.2): lanes 0-15 rows 0-15 hold K {0..7,16..23},
// lanes 16-31 same rows hold K {8..15,24..31}.
__device__ inline v16bf load_fragA(const unsigned short* base, int stride, int lane) {
  const unsigned short* p = base + (size_t)(lane & 15) * stride + ((lane >> 4) << 3);
  FragU t;
  t.u[0] = *(const uint4*)(p);
  t.u[1] = *(const uint4*)(p + 16);
  return t.v;
}
// B-matrix 32x16 bf16 fragment: lane (n = lane&15) holds column n; half-wave 0 holds
// K {0..7,16..23}, half-wave 1 holds K {8..15,24..31}. Source is row-major B^T.
__device__ inline v16bf load_fragB(const unsigned short* base, int stride, int lane) {
  const unsigned short* p = base + (size_t)(lane & 15) * stride + ((lane >> 4) << 3);
  FragU t;
  t.u[0] = *(const uint4*)(p);
  t.u[1] = *(const uint4*)(p + 16);
  return t.v;
}

__device__ inline v8f wmma_bf(v16bf a, v16bf b, v8f c) {
  // D = A*B + C, fp32 accumulate
  return __builtin_amdgcn_wmma_f32_16x16x32_bf16(false, a, false, b, (short)0, c,
                                                 false, false);
}

// ---------------------------------------------------------------------------
// Weight fp32 -> bf16 conversion with optional row/col zero-padding
// ---------------------------------------------------------------------------
__global__ __launch_bounds__(256) void cvt_bf16_pad(
    const float* __restrict__ src, unsigned short* __restrict__ dst,
    int rows_real, int cols_real, int cols_pad, int total) {
  int idx = blockIdx.x * 256 + threadIdx.x;
  if (idx >= total) return;
  int r = idx / cols_pad;
  int ccol = idx - r * cols_pad;
  float v = 0.f;
  if (r < rows_real && ccol < cols_real) v = src[(size_t)r * cols_real + ccol];
  dst[idx] = f2bf(v);
}

// ---------------------------------------------------------------------------
// adaLN: ada[b, o] = bias[o] + sum_k silu(c[b,k]) * w[o,k]   (B=4, tiny GEMM)
// ---------------------------------------------------------------------------
__global__ __launch_bounds__(256) void ada_kernel(
    const float* __restrict__ c, const float* __restrict__ w,
    const float* __restrict__ bias, float* __restrict__ ada) {
  __shared__ float cs[CDIM];
  int idx = blockIdx.x * 256 + threadIdx.x;   // blocks never straddle b (6144%256==0)
  int b = idx / ADA6;
  int o = idx - b * ADA6;
  for (int i = 0; i < 4; i++) {
    int kk = threadIdx.x + i * 256;
    float cv = c[(size_t)b * CDIM + kk];
    cs[kk] = cv / (1.f + __expf(-cv));
  }
  __syncthreads();
  float acc = bias[o];
  const float* wr = w + (size_t)o * CDIM;
  for (int kk = 0; kk < CDIM; kk++) acc += cs[kk] * wr[kk];
  ada[(size_t)b * ADA6 + o] = acc;
}

// ---------------------------------------------------------------------------
// RMSNorm + adaLN modulate -> bf16.  One block per token row.
// ---------------------------------------------------------------------------
__global__ __launch_bounds__(256) void rmsnorm_mod(
    const float* __restrict__ x, const float* __restrict__ w,
    const float* __restrict__ ada, int shift_off, int scale_off,
    unsigned short* __restrict__ out) {
  __shared__ float red[256];
  int row = blockIdx.x;
  int tid = threadIdx.x;
  int b = row >> 10;
  float xv[4];
  float ss = 0.f;
  for (int i = 0; i < 4; i++) {
    xv[i] = x[(size_t)row * CDIM + tid + i * 256];
    ss += xv[i] * xv[i];
  }
  red[tid] = ss;
  __syncthreads();
  for (int s = 128; s > 0; s >>= 1) {
    if (tid < s) red[tid] += red[tid + s];
    __syncthreads();
  }
  float scale = rsqrtf(red[0] * (1.f / CDIM) + 1e-6f);
  for (int i = 0; i < 4; i++) {
    int ch = tid + i * 256;
    float g  = ada[(size_t)b * ADA6 + scale_off + ch];
    float sf = ada[(size_t)b * ADA6 + shift_off + ch];
    out[(size_t)row * CDIM + ch] = f2bf((xv[i] * scale * w[ch]) * (1.f + g) + sf);
  }
}

// ---------------------------------------------------------------------------
// Generic bf16 WMMA GEMM:  OUT[m,n] = sum_k A[m,k]*W[n,k] (+ bias[n]) + epilogue
// Block 256 thr = 8 waves; block tile 256x128; wave tile 64x64 (4x4 frags,
// 16 WMMAs per 16 b128 loads per K-step -> 1.0 WMMA per b128).
// epi 0: bf16 store to outB (guard n<Nreal)                      [w12]
// epi 1: scatter into q (B,H,N,D), k (B,H,N,D), v^T (B,H,D,N)    [qkv]
// epi 2: outF = resid + ada[b, gate_off+n] * (acc+bias)          [proj, w3]
// ---------------------------------------------------------------------------
__global__ __launch_bounds__(256) void gemm_bf16(
    const unsigned short* __restrict__ A, int lda,
    const unsigned short* __restrict__ W, int ldw,
    const float* __restrict__ bias,
    int K, int Nreal, int epi,
    float* __restrict__ outF, unsigned short* __restrict__ outB, int ldo,
    const float* __restrict__ resid, const float* __restrict__ ada, int gate_off,
    unsigned short* __restrict__ qd, unsigned short* __restrict__ kd,
    unsigned short* __restrict__ vd) {
  const int lane  = threadIdx.x & 31;
  const int wv    = threadIdx.x >> 5;
  const int mbase = blockIdx.y * 256 + (wv & 3) * 64;   // 4 waves tile M
  const int nbase = blockIdx.x * 128 + (wv >> 2) * 64;  // 2 waves tile N

  v8f acc[4][4];
  for (int i = 0; i < 4; i++)
    for (int j = 0; j < 4; j++)
      for (int e = 0; e < 8; e++) acc[i][j][e] = 0.f;

  for (int k0 = 0; k0 < K; k0 += 32) {
    v16bf af[4], wf[4];
    for (int i = 0; i < 4; i++)
      af[i] = load_fragA(A + (size_t)(mbase + i * 16) * lda + k0, lda, lane);
    for (int j = 0; j < 4; j++)
      wf[j] = load_fragB(W + (size_t)(nbase + j * 16) * ldw + k0, ldw, lane);
    for (int i = 0; i < 4; i++)
      for (int j = 0; j < 4; j++)
        acc[i][j] = wmma_bf(af[i], wf[j], acc[i][j]);
  }

  // C/D layout: lane<16 -> (M = v, N = lane); lane>=16 -> (M = 8+v, N = lane-16)
  const int half = lane >> 4;
  const int nl   = lane & 15;
  for (int i = 0; i < 4; i++) {
    for (int j = 0; j < 4; j++) {
      int n_g = nbase + j * 16 + nl;
      float bv = (n_g < Nreal) ? bias[n_g] : 0.f;
      for (int v = 0; v < 8; v++) {
        int m_g = mbase + i * 16 + half * 8 + v;
        float val = acc[i][j][v] + bv;
        if (epi == 0) {
          if (n_g < Nreal) outB[(size_t)m_g * ldo + n_g] = f2bf(val);
        } else if (epi == 1) {
          int which = n_g >> 10;         // 0=q 1=k 2=v
          int hh    = (n_g & 1023) >> 6; // head
          int dd    = n_g & 63;          // dim
          int b     = m_g >> 10;
          int nn    = m_g & 1023;
          unsigned short bvv = f2bf(val);
          if (which == 2) {
            vd[((size_t)(b * HNUM + hh) * DDIM + dd) * NSEQ + nn] = bvv; // V^T
          } else {
            size_t qk = ((size_t)(b * HNUM + hh) * NSEQ + nn) * DDIM + dd;
            if (which == 0) qd[qk] = bvv; else kd[qk] = bvv;
          }
        } else {
          int b = m_g >> 10;
          float g = ada[(size_t)b * ADA6 + gate_off + n_g];
          outF[(size_t)m_g * ldo + n_g] = resid[(size_t)m_g * ldo + n_g] + g * val;
        }
      }
    }
  }
}

// ---------------------------------------------------------------------------
// q/k per-head RMSNorm over D=64 (in place, bf16).
// The softmax 1/sqrt(D) scale is folded into the q-side weight (free here).
// ---------------------------------------------------------------------------
__global__ __launch_bounds__(256) void qk_norm(
    unsigned short* __restrict__ q, unsigned short* __restrict__ k,
    const float* __restrict__ qw, const float* __restrict__ kw) {
  int idx = blockIdx.x * 256 + threadIdx.x;   // 2 * B*H*N rows
  int isK = idx >> 16;
  int row = idx & 65535;
  unsigned short* p = (isK ? k : q) + (size_t)row * DDIM;
  const float* wn = isK ? kw : qw;
  float post = isK ? 1.0f : 0.125f;           // 1/sqrt(64) folded into q
  float vals[DDIM];
  float ss = 0.f;
  for (int d = 0; d < DDIM; d++) { vals[d] = bf2f(p[d]); ss += vals[d] * vals[d]; }
  float r = rsqrtf(ss * (1.f / DDIM) + 1e-6f) * post;
  for (int d = 0; d < DDIM; d++) p[d] = f2bf(vals[d] * r * wn[d]);
}

// ---------------------------------------------------------------------------
// Flash attention per (b,h): grid (N/128, B*H), 8 waves, wave owns 16 q rows.
// Keys in chunks of 32: S(16x32) via 4 WMMAs, online softmax (half-wave
// shfl reductions), P -> LDS -> A-frag, O(16x64) += P @ V via 4 WMMAs.
// ---------------------------------------------------------------------------
__global__ __launch_bounds__(256) void attn_kernel(
    const unsigned short* __restrict__ q, const unsigned short* __restrict__ k,
    const unsigned short* __restrict__ vt, unsigned short* __restrict__ out) {
  __shared__ __align__(16) unsigned short plds[8][16][32];
  const int lane  = threadIdx.x & 31;
  const int wv    = threadIdx.x >> 5;
  const int bh    = blockIdx.y;
  const int qbase = blockIdx.x * 128 + wv * 16;
  const int half  = lane >> 4;
  const int nl    = lane & 15;

  const unsigned short* qh = q  + (size_t)bh * NSEQ * DDIM;
  const unsigned short* kh = k  + (size_t)bh * NSEQ * DDIM;
  const unsigned short* vh = vt + (size_t)bh * DDIM * NSEQ;

  v16bf qa[2];
  for (int kk = 0; kk < 2; kk++)
    qa[kk] = load_fragA(qh + (size_t)qbase * DDIM + kk * 32, DDIM, lane);

  float mrow[8], lrow[8];
  v8f o[4];
  for (int v = 0; v < 8; v++) { mrow[v] = -1e30f; lrow[v] = 0.f; }
  for (int t = 0; t < 4; t++)
    for (int v = 0; v < 8; v++) o[t][v] = 0.f;

  for (int jc = 0; jc < NSEQ / 32; jc++) {
    int kbase = jc * 32;
    v8f s[2];
    for (int t = 0; t < 2; t++)
      for (int v = 0; v < 8; v++) s[t][v] = 0.f;
    for (int kk = 0; kk < 2; kk++)
      for (int t = 0; t < 2; t++) {
        v16bf kb = load_fragB(kh + (size_t)(kbase + t * 16) * DDIM + kk * 32,
                              DDIM, lane);
        s[t] = wmma_bf(qa[kk], kb, s[t]);
      }

    float cm[8], p0[8], p1[8], alpha[8], rs[8];
    for (int v = 0; v < 8; v++) cm[v] = fmaxf(s[0][v], s[1][v]);
    for (int off = 1; off < 16; off <<= 1)     // reduce within 16-lane row group
      for (int v = 0; v < 8; v++)
        cm[v] = fmaxf(cm[v], __shfl_xor(cm[v], off, 32));
    for (int v = 0; v < 8; v++) {
      float nm = fmaxf(mrow[v], cm[v]);
      alpha[v] = __expf(mrow[v] - nm);
      p0[v] = __expf(s[0][v] - nm);
      p1[v] = __expf(s[1][v] - nm);
      rs[v] = p0[v] + p1[v];
      mrow[v] = nm;
    }
    for (int off = 1; off < 16; off <<= 1)
      for (int v = 0; v < 8; v++)
        rs[v] += __shfl_xor(rs[v], off, 32);
    for (int v = 0; v < 8; v++) lrow[v] = lrow[v] * alpha[v] + rs[v];
    for (int t = 0; t < 4; t++)
      for (int v = 0; v < 8; v++) o[t][v] *= alpha[v];

    // P (C-layout) -> LDS -> A-layout fragment
    for (int v = 0; v < 8; v++) {
      plds[wv][half * 8 + v][nl]      = f2bf(p0[v]);
      plds[wv][half * 8 + v][16 + nl] = f2bf(p1[v]);
    }
    __syncthreads();
    v16bf pa = load_fragA(&plds[wv][0][0], 32, lane);
    __syncthreads();

    for (int t = 0; t < 4; t++) {
      v16bf vb = load_fragB(vh + (size_t)(t * 16) * NSEQ + kbase, NSEQ, lane);
      o[t] = wmma_bf(pa, vb, o[t]);
    }
  }

  int b = bh >> 4;
  int h = bh & 15;
  for (int v = 0; v < 8; v++) {
    float inv = 1.f / lrow[v];
    int nrow = qbase + half * 8 + v;
    size_t rowoff = ((size_t)(b * NSEQ + nrow)) * CDIM + h * DDIM;
    for (int t = 0; t < 4; t++)
      out[rowoff + t * 16 + nl] = f2bf(o[t][v] * inv);
  }
}

// ---------------------------------------------------------------------------
// SwiGLU combine: h[m,f] = silu(x1)*x2 for f<FFH, 0 in padded tail
// ---------------------------------------------------------------------------
__global__ __launch_bounds__(256) void swiglu_comb(
    const unsigned short* __restrict__ raw, unsigned short* __restrict__ h) {
  int idx = blockIdx.x * 256 + threadIdx.x;   // BN_TOK * FFHP exactly
  int m = idx / FFHP;
  int f = idx - m * FFHP;
  float val = 0.f;
  if (f < FFH) {
    float x1 = bf2f(raw[(size_t)m * FF2P + f]);
    float x2 = bf2f(raw[(size_t)m * FF2P + FFH + f]);
    val = x1 / (1.f + __expf(-x1)) * x2;
  }
  h[(size_t)m * FFHP + f] = f2bf(val);
}

// ---------------------------------------------------------------------------
extern "C" void kernel_launch(void* const* d_in, const int* in_sizes, int n_in,
                              void* d_out, int out_size, void* d_ws, size_t ws_size,
                              hipStream_t stream) {
  const float* x     = (const float*)d_in[0];
  const float* c     = (const float*)d_in[1];
  const float* n1w   = (const float*)d_in[2];
  const float* n2w   = (const float*)d_in[3];
  const float* qkvw  = (const float*)d_in[4];
  const float* qkvb  = (const float*)d_in[5];
  const float* qnw   = (const float*)d_in[6];
  const float* knw   = (const float*)d_in[7];
  const float* projw = (const float*)d_in[8];
  const float* projb = (const float*)d_in[9];
  const float* w12w  = (const float*)d_in[10];
  const float* w12b  = (const float*)d_in[11];
  const float* w3w   = (const float*)d_in[12];
  const float* w3b   = (const float*)d_in[13];
  const float* adaw  = (const float*)d_in[14];
  const float* adab  = (const float*)d_in[15];
  float* out = (float*)d_out;

  char* ws = (char*)d_ws;
  size_t off = 0;
  auto alloc = [&](size_t bytes) -> char* {
    char* p = ws + off;
    off += (bytes + 255) & ~(size_t)255;
    return p;
  };
  float*          ada      = (float*)         alloc((size_t)BDIM * ADA6 * 4);
  unsigned short* xm1      = (unsigned short*)alloc((size_t)BN_TOK * CDIM * 2);
  unsigned short* qkvw_bf  = (unsigned short*)alloc((size_t)3 * CDIM * CDIM * 2);
  unsigned short* projw_bf = (unsigned short*)alloc((size_t)CDIM * CDIM * 2);
  unsigned short* w12w_bf  = (unsigned short*)alloc((size_t)FF2P * CDIM * 2);
  unsigned short* w3w_bf   = (unsigned short*)alloc((size_t)CDIM * FFHP * 2);
  unsigned short* qbuf     = (unsigned short*)alloc((size_t)BN_TOK * CDIM * 2);
  unsigned short* kbuf     = (unsigned short*)alloc((size_t)BN_TOK * CDIM * 2);
  unsigned short* vtbuf    = (unsigned short*)alloc((size_t)BN_TOK * CDIM * 2);
  unsigned short* aobuf    = (unsigned short*)alloc((size_t)BN_TOK * CDIM * 2);
  float*          x1       = (float*)         alloc((size_t)BN_TOK * CDIM * 4);
  unsigned short* xm2      = (unsigned short*)alloc((size_t)BN_TOK * CDIM * 2);
  unsigned short* w12raw   = (unsigned short*)alloc((size_t)BN_TOK * FF2P * 2);
  unsigned short* hbuf     = (unsigned short*)alloc((size_t)BN_TOK * FFHP * 2);

  // 1) weights -> bf16 (zero-padded where tiles need it)
  cvt_bf16_pad<<<(3 * CDIM * CDIM + 255) / 256, 256, 0, stream>>>(
      qkvw, qkvw_bf, 3 * CDIM, CDIM, CDIM, 3 * CDIM * CDIM);
  cvt_bf16_pad<<<(CDIM * CDIM + 255) / 256, 256, 0, stream>>>(
      projw, projw_bf, CDIM, CDIM, CDIM, CDIM * CDIM);
  cvt_bf16_pad<<<(FF2P * CDIM + 255) / 256, 256, 0, stream>>>(
      w12w, w12w_bf, FF2, CDIM, CDIM, FF2P * CDIM);
  cvt_bf16_pad<<<(CDIM * FFHP + 255) / 256, 256, 0, stream>>>(
      w3w, w3w_bf, CDIM, FFH, FFHP, CDIM * FFHP);

  // 2) adaLN vectors
  ada_kernel<<<BDIM * ADA6 / 256, 256, 0, stream>>>(c, adaw, adab, ada);

  // 3) norm1 + modulate(s1,g1)
  rmsnorm_mod<<<BN_TOK, 256, 0, stream>>>(x, n1w, ada, 0, CDIM, xm1);

  // 4) QKV GEMM, scatter to q/k/v^T   (M=4096 -> 16 blocks of 256)
  gemm_bf16<<<dim3(24, 16), 256, 0, stream>>>(
      xm1, CDIM, qkvw_bf, CDIM, qkvb, CDIM, 3 * CDIM, 1,
      nullptr, nullptr, 0, nullptr, nullptr, 0, qbuf, kbuf, vtbuf);

  // 5) q/k head RMSNorm (+1/sqrt(D) fold into q)
  qk_norm<<<(2 * BDIM * HNUM * NSEQ) / 256, 256, 0, stream>>>(qbuf, kbuf, qnw, knw);

  // 6) attention
  attn_kernel<<<dim3(NSEQ / 128, BDIM * HNUM), 256, 0, stream>>>(
      qbuf, kbuf, vtbuf, aobuf);

  // 7) proj GEMM + b1-gated residual -> x1
  gemm_bf16<<<dim3(8, 16), 256, 0, stream>>>(
      aobuf, CDIM, projw_bf, CDIM, projb, CDIM, CDIM, 2,
      x1, nullptr, CDIM, x, ada, 2 * CDIM, nullptr, nullptr, nullptr);

  // 8) norm2 + modulate(s2,g2)
  rmsnorm_mod<<<BN_TOK, 256, 0, stream>>>(x1, n2w, ada, 3 * CDIM, 4 * CDIM, xm2);

  // 9) w12 GEMM (padded N) -> raw bf16
  gemm_bf16<<<dim3(FF2P / 128, 16), 256, 0, stream>>>(
      xm2, CDIM, w12w_bf, CDIM, w12b, CDIM, FF2, 0,
      nullptr, w12raw, FF2P, nullptr, nullptr, 0, nullptr, nullptr, nullptr);

  // 10) SwiGLU combine (zero-fills padded K tail)
  swiglu_comb<<<(BN_TOK * FFHP) / 256, 256, 0, stream>>>(w12raw, hbuf);

  // 11) w3 GEMM + b2-gated residual -> output
  gemm_bf16<<<dim3(8, 16), 256, 0, stream>>>(
      hbuf, FFHP, w3w_bf, FFHP, w3b, FFHP, CDIM, 2,
      out, nullptr, CDIM, x1, ada, 5 * CDIM, nullptr, nullptr, nullptr);
}
